// HSTUTransducer_566935683422
// MI455X (gfx1250) — matrix operand order repaired and checked by reading, not compiled, not run on hardware
//
#include <hip/hip_runtime.h>
#include <hip/hip_bf16.h>
#include <math.h>

// ---------------------------------------------------------------------------
// HSTU transducer forward for MI455X (gfx1250, wave32, WMMA bf16 path).
// Only row `last[b]` of seq_out is consumed by the head, so attention /
// LN2 / out-proj run on one 16-row query tile per (b,h).
// Round 2: 16x64 register-blocked uvqk GEMM (16 WMMA/wave, 4x A reuse),
// split-K attention (8 key-splits per (b,h)) + partial-O reduction.
// ---------------------------------------------------------------------------

typedef __bf16 bf16_t;
typedef __attribute__((ext_vector_type(16))) __bf16 v16bf;
typedef __attribute__((ext_vector_type(8)))  __bf16 v8bf;
typedef __attribute__((ext_vector_type(8)))  float  v8f;

#define BB 8
#define SS 2048
#define DD 128
#define HH 4
#define TH 512           // H*DH
#define TA 512           // H*DA
#define NE 2048          // 2*TH + 2*TA
#define VV 100000
#define NPOS 256
#define NTIME 256
#define EPS 1e-6f
#define NSPLIT 8         // attention key splits
#define KCHUNK (SS / NSPLIT)   // 256 keys per split

__device__ __forceinline__ v8f wmma_bf16(v16bf a, v16bf b, v8f c) {
  // v_wmma_f32_16x16x32_bf16: D = A(16x32) * B(32x16) + C(16x16 f32)
  return __builtin_amdgcn_wmma_f32_16x16x32_bf16(
      false, a, false, b, (short)0, c, false, false);
}

// A-matrix 16x32 bf16 fragment: lane&15 = M row.
// lanes 0-15 hold K = k0..k0+15; lanes 16-31 K = k0+16..k0+31.
__device__ __forceinline__ v16bf load_afrag(const bf16_t* __restrict__ rowp,
                                            int k0, int lane) {
  int kb = (lane < 16) ? 0 : 16;
  v8bf lo = *(const v8bf*)(rowp + k0 + kb);
  v8bf hi = *(const v8bf*)(rowp + k0 + kb + 8);
  return __builtin_shufflevector(lo, hi, 0, 1, 2, 3, 4, 5, 6, 7,
                                 8, 9, 10, 11, 12, 13, 14, 15);
}

__device__ __forceinline__ float silu(float x) {
  return x / (1.0f + __expf(-x));
}

// ---------------------------------------------------------------------------
// Kernel 0: per-batch meta {seq_len, high, last, qt}
// ---------------------------------------------------------------------------
__global__ void prep_kernel(const int* __restrict__ seq_lengths,
                            const int* __restrict__ num_targets,
                            const int* __restrict__ timestamps,
                            int4* __restrict__ meta) {
  int b = threadIdx.x;
  if (b < BB) {
    int sl   = seq_lengths[b];
    int nt   = num_targets[b];
    int last = min(max(sl - 1, 0), SS - 1);
    meta[b]  = make_int4(sl, sl - nt, last, timestamps[b * SS + last]);
  }
}

// ---------------------------------------------------------------------------
// Kernel 1: uvqk_w [128,2048] f32 -> wT [2048,128] bf16 (K contiguous)
// ---------------------------------------------------------------------------
__global__ void convert_wT_kernel(const float* __restrict__ w,
                                  bf16_t* __restrict__ wT) {
  int i = blockIdx.x * blockDim.x + threadIdx.x;
  if (i < NE * DD) {
    int c = i >> 7, kd = i & (DD - 1);
    wT[i] = (bf16_t)w[kd * NE + c];
  }
}

// ---------------------------------------------------------------------------
// Kernel 2: embedding + pos + time, fused LayerNorm -> nx bf16 [B*S,128].
// Also stores x (pre-LN) at the `last` row of each batch.
// ---------------------------------------------------------------------------
__global__ void embed_ln_kernel(const float* __restrict__ emb,
                                const float* __restrict__ pos_emb,
                                const float* __restrict__ time_emb,
                                const float* __restrict__ g,
                                const float* __restrict__ beta,
                                const int*  __restrict__ item_ids,
                                const int*  __restrict__ timestamps,
                                const int4* __restrict__ meta,
                                bf16_t* __restrict__ nx,
                                float*  __restrict__ x_last) {
  int token = blockIdx.x;
  int bb = token >> 11, s = token & (SS - 1);
  int d = threadIdx.x;                       // 0..127
  int4 mt = meta[bb];
  int high = mt.y, last = mt.z, qt = mt.w;

  int col  = min(s, high);                   // clip(s, 0, high); s >= 0
  int pcol = min(max(high - col, 0), NPOS - 1);
  int t = timestamps[token];
  float dt = fmaxf((float)(qt - t), 1e-6f) / 60.0f;
  float fs = fminf(sqrtf(dt), (float)NTIME);
  int ti = (int)fs;                          // 0..256 (NTIME+1 rows)
  int item = item_ids[token];

  float x = emb[(size_t)item * DD + d] * 11.313708499f   // sqrt(128)
          + pos_emb[pcol * DD + d] + time_emb[ti * DD + d];

  __shared__ float red[DD];
  red[d] = x;
  __syncthreads();
  for (int off = 64; off > 0; off >>= 1) {
    if (d < off) red[d] += red[d + off];
    __syncthreads();
  }
  float mu = red[0] * (1.0f / DD);
  __syncthreads();
  float dv = x - mu;
  red[d] = dv * dv;
  __syncthreads();
  for (int off = 64; off > 0; off >>= 1) {
    if (d < off) red[d] += red[d + off];
    __syncthreads();
  }
  float rstd = rsqrtf(red[0] * (1.0f / DD) + EPS);

  nx[(size_t)token * DD + d] = (bf16_t)((x - mu) * rstd * g[d] + beta[d]);
  if (s == last) x_last[bb * DD + d] = x;
}

// ---------------------------------------------------------------------------
// Kernel 3: uvqk GEMM  C[16384,2048] = nx_bf16 @ wT_bf16 + bias
// Register-blocked: each wave computes a 16(M) x 64(N) strip -> 16 WMMA,
// each A-fragment loaded once and reused across 4 B-fragments.
// Column strips (64 wide) never straddle the u/v/q/k 512-col boundaries.
// Epilogue scatters:
//   c in [0,512)    : silu -> u_last (only at s == last[b])
//   c in [512,1024) : v  -> vT  [B,H,128,S]  bf16 (transposed for P@V)
//   c in [1024,1536): q  -> q   [B,H,S,128]  bf16
//   c in [1536,2048): k  -> k   [B,H,S,128]  bf16
// ---------------------------------------------------------------------------
__global__ void uvqk_gemm_kernel(const bf16_t* __restrict__ nx,
                                 const bf16_t* __restrict__ wT,
                                 const float*  __restrict__ bias,
                                 const int4*   __restrict__ meta,
                                 float*  __restrict__ u_last,
                                 bf16_t* __restrict__ qbuf,
                                 bf16_t* __restrict__ kbuf,
                                 bf16_t* __restrict__ vT) {
  int lane   = threadIdx.x & 31;
  int strip_n = blockIdx.x;                           // 0..31  (64-col strips)
  int tile_m  = blockIdx.y * blockDim.y + threadIdx.y; // 0..1023

  int n = lane & 15;
  int khalf = (lane < 16) ? 0 : 16;
  const bf16_t* arow = nx + (size_t)(tile_m * 16 + n) * DD;

  v8f acc[4];
#pragma unroll
  for (int nb = 0; nb < 4; ++nb)
    acc[nb] = (v8f){0.f, 0.f, 0.f, 0.f, 0.f, 0.f, 0.f, 0.f};

#pragma unroll
  for (int kk = 0; kk < 4; ++kk) {
    v16bf a = load_afrag(arow, kk * 32, lane);        // loaded once, used 4x
#pragma unroll
    for (int nb = 0; nb < 4; ++nb) {
      const bf16_t* brow =
          wT + (size_t)(strip_n * 64 + nb * 16 + n) * DD + khalf + kk * 32;
      v16bf b = *(const v16bf*)brow;                  // B 32x16, K contiguous
      acc[nb] = wmma_bf16(a, b, acc[nb]);
    }
  }

  int mbase = (lane < 16) ? 0 : 8;
#pragma unroll
  for (int nb = 0; nb < 4; ++nb) {
    int c = strip_n * 64 + nb * 16 + n;
    float bv = bias[c];
#pragma unroll
    for (int j = 0; j < 8; ++j) {
      int r = tile_m * 16 + mbase + j;
      int bb = r >> 11, s = r & (SS - 1);
      float val = acc[nb][j] + bv;
      if (c < TH) {                                   // u (only needed at last)
        if (s == meta[bb].z) u_last[bb * TH + c] = silu(val);
      } else if (c < TH + TA) {                       // v -> transposed
        int c2 = c - TH, h = c2 >> 7, d = c2 & 127;
        vT[(((size_t)(bb * HH + h)) * DD + d) * SS + s] = (bf16_t)val;
      } else if (c < TH + 2 * TA) {                   // q
        int c2 = c - (TH + TA), h = c2 >> 7, d = c2 & 127;
        qbuf[(((size_t)(bb * HH + h)) * SS + s) * DD + d] = (bf16_t)val;
      } else {                                        // k
        int c2 = c - (TH + 2 * TA), h = c2 >> 7, d = c2 & 127;
        kbuf[(((size_t)(bb * HH + h)) * SS + s) * DD + d] = (bf16_t)val;
      }
    }
  }
}

// ---------------------------------------------------------------------------
// Kernel 4: attention, split-K. Grid (B*H, NSPLIT), one wave per block.
// Each wave accumulates partial O over a 256-key chunk for the 16-row
// query tile containing last[b]; silu attention is linear in V, so the
// partials sum exactly. Scores: 8 WMMA per 32 keys; P relayout via LDS
// (C-layout -> A-layout); P@V: 8 WMMA into 8 resident f32 accumulators.
// ---------------------------------------------------------------------------
__global__ void attn_kernel(const bf16_t* __restrict__ qbuf,
                            const bf16_t* __restrict__ kbuf,
                            const bf16_t* __restrict__ vT,
                            const int4*   __restrict__ meta,
                            float* __restrict__ o_part) {
  int bh = blockIdx.x;                 // 0..31
  int split = blockIdx.y;              // 0..NSPLIT-1
  int bb = bh >> 2;
  int lane = threadIdx.x;              // 0..31, one wave
  int4 mt = meta[bb];
  int sl = mt.x, last = mt.z;
  int q0 = last & ~15;

  const bf16_t* qb = qbuf + ((size_t)bh * SS) * DD;
  const bf16_t* kb = kbuf + ((size_t)bh * SS) * DD;
  const bf16_t* vb = vT   + ((size_t)bh * DD) * SS;

  int m = lane & 15;
  int mbase = (lane < 16) ? 0 : 8;
  int n = lane & 15;
  int khalf = (lane < 16) ? 0 : 16;

  v8f oacc[8];
#pragma unroll
  for (int c = 0; c < 8; ++c)
    oacc[c] = (v8f){0.f, 0.f, 0.f, 0.f, 0.f, 0.f, 0.f, 0.f};

  int key_begin = split * KCHUNK;
  int key_end   = min(last + 1, key_begin + KCHUNK);  // exclusive

  if (key_begin < key_end) {
    v16bf qa[4];
#pragma unroll
    for (int kk = 0; kk < 4; ++kk)
      qa[kk] = load_afrag(qb + (size_t)(q0 + m) * DD, kk * 32, lane);

    __shared__ bf16_t ptile[16][32];
    const float scale = 0.08838834764f;  // DA^-0.5

    for (int key0 = key_begin; key0 < key_end; key0 += 32) {
#pragma unroll
      for (int t = 0; t < 2; ++t) {
        v8f sacc = {0.f, 0.f, 0.f, 0.f, 0.f, 0.f, 0.f, 0.f};
        const bf16_t* krow = kb + (size_t)(key0 + t * 16 + n) * DD + khalf;
#pragma unroll
        for (int kk = 0; kk < 4; ++kk) {
          v16bf bf = *(const v16bf*)(krow + kk * 32);
          sacc = wmma_bf16(qa[kk], bf, sacc);
        }
        int key = key0 + t * 16 + n;
        bool kvalid = (key < sl);
#pragma unroll
        for (int j = 0; j < 8; ++j) {
          int qrow = q0 + mbase + j;
          float sv = sacc[j] * scale;
          float p = (kvalid && key <= qrow) ? silu(sv) : 0.0f;
          ptile[mbase + j][t * 16 + n] = (bf16_t)p;
        }
      }
      __syncthreads();
      // P as A-matrix 16x32 fragment
      v16bf pa;
      {
        int kbase = (lane < 16) ? 0 : 16;
#pragma unroll
        for (int i = 0; i < 8; ++i) {
          pa[i]     = ptile[m][kbase + i];
          pa[8 + i] = ptile[m][kbase + 8 + i];
        }
      }
      __syncthreads();
#pragma unroll
      for (int c = 0; c < 8; ++c) {
        const bf16_t* vrow = vb + (size_t)(c * 16 + n) * SS + key0 + khalf;
        v16bf vf = *(const v16bf*)vrow;
        oacc[c] = wmma_bf16(pa, vf, oacc[c]);
      }
    }
  }

  // store only row `last` of the tile: M = j + 8*(lane>=16)
  int mt_row = last & 15;
  if ((lane >> 4) == (mt_row >> 3)) {
    int j = mt_row & 7;
#pragma unroll
    for (int c = 0; c < 8; ++c)
      o_part[((size_t)(bh * NSPLIT + split)) * DD + c * 16 + n] = oacc[c][j];
  }
}

// ---------------------------------------------------------------------------
// Kernel 4b: reduce split-K partials: o_last[b*512+h*128+d] = sum partials
// ---------------------------------------------------------------------------
__global__ void attn_reduce_kernel(const float* __restrict__ o_part,
                                   float* __restrict__ o_last) {
  int i = blockIdx.x * blockDim.x + threadIdx.x;   // over BB*TH = 4096
  if (i < BB * TH) {
    int bh = i >> 7;                               // b*HH + h
    int d  = i & (DD - 1);
    float s = 0.f;
#pragma unroll
    for (int sp = 0; sp < NSPLIT; ++sp)
      s += o_part[((size_t)(bh * NSPLIT + sp)) * DD + d];
    o_last[i] = s;
  }
}

// ---------------------------------------------------------------------------
// Kernel 5: tail for the 8 last rows: LN2, y = [u, x, u*no], h = y@out_w + x
// ---------------------------------------------------------------------------
__global__ void tail_kernel(const float* __restrict__ o_last,
                            const float* __restrict__ u_last,
                            const float* __restrict__ x_last,
                            const float* __restrict__ ln2_g,
                            const float* __restrict__ ln2_b,
                            const float* __restrict__ out_w,
                            float* __restrict__ hbuf) {
  int bb = blockIdx.x;
  int tid = threadIdx.x;               // 128 threads
  __shared__ float y[2 * TH + DD];     // 1152
  __shared__ float red[DD];
  const float* o = o_last + bb * TH;

  float ps = 0.f;
  for (int i = tid; i < TH; i += DD) ps += o[i];
  red[tid] = ps;
  __syncthreads();
  for (int off = 64; off > 0; off >>= 1) {
    if (tid < off) red[tid] += red[tid + off];
    __syncthreads();
  }
  float mu = red[0] * (1.0f / TH);
  __syncthreads();
  float pv = 0.f;
  for (int i = tid; i < TH; i += DD) { float d = o[i] - mu; pv += d * d; }
  red[tid] = pv;
  __syncthreads();
  for (int off = 64; off > 0; off >>= 1) {
    if (tid < off) red[tid] += red[tid + off];
    __syncthreads();
  }
  float rstd = rsqrtf(red[0] * (1.0f / TH) + EPS);
  __syncthreads();

  for (int i = tid; i < TH; i += DD) {
    float no = (o[i] - mu) * rstd * ln2_g[i] + ln2_b[i];
    float u = u_last[bb * TH + i];
    y[i] = u;
    y[TH + DD + i] = u * no;
  }
  y[TH + tid] = x_last[bb * DD + tid];
  __syncthreads();

  float acc = x_last[bb * DD + tid];   // residual
  for (int f = 0; f < 2 * TH + DD; ++f)
    acc += y[f] * out_w[f * DD + tid];
  hbuf[bb * DD + tid] = acc;
}

// ---------------------------------------------------------------------------
// Kernel 6: head, out[b,v] = h[b,:] . head_w[:,v] + head_b[v]
// Bandwidth-bound on head_w (51 MB); coalesced over v.
// ---------------------------------------------------------------------------
__global__ void head_kernel(const float* __restrict__ hbuf,
                            const float* __restrict__ head_w,
                            const float* __restrict__ head_b,
                            float* __restrict__ out) {
  __shared__ float hs[BB * DD];
  for (int i = threadIdx.x; i < BB * DD; i += blockDim.x) hs[i] = hbuf[i];
  __syncthreads();
  int v = blockIdx.x * blockDim.x + threadIdx.x;
  if (v >= VV) return;
  float acc[BB];
#pragma unroll
  for (int b = 0; b < BB; ++b) acc[b] = 0.f;
  for (int d = 0; d < DD; ++d) {
    float w = head_w[(size_t)d * VV + v];
#pragma unroll
    for (int b = 0; b < BB; ++b) acc[b] += hs[b * DD + d] * w;
  }
  float hb = head_b[v];
#pragma unroll
  for (int b = 0; b < BB; ++b) out[(size_t)b * VV + v] = acc[b] + hb;
}

// ---------------------------------------------------------------------------
extern "C" void kernel_launch(void* const* d_in, const int* in_sizes, int n_in,
                              void* d_out, int out_size, void* d_ws, size_t ws_size,
                              hipStream_t stream) {
  (void)in_sizes; (void)n_in; (void)out_size; (void)ws_size;
  const float* emb_table  = (const float*)d_in[0];
  const float* pos_emb    = (const float*)d_in[1];
  const float* time_emb   = (const float*)d_in[2];
  const float* ln1_g      = (const float*)d_in[3];
  const float* ln1_b      = (const float*)d_in[4];
  const float* uvqk_w     = (const float*)d_in[5];
  const float* uvqk_b     = (const float*)d_in[6];
  const float* ln2_g      = (const float*)d_in[7];
  const float* ln2_b      = (const float*)d_in[8];
  const float* out_w      = (const float*)d_in[9];
  const float* head_w     = (const float*)d_in[10];
  const float* head_b     = (const float*)d_in[11];
  const int*   item_ids   = (const int*)d_in[12];
  const int*   seq_len    = (const int*)d_in[13];
  const int*   timestamps = (const int*)d_in[14];
  const int*   num_tgt    = (const int*)d_in[15];
  float* out = (float*)d_out;

  // workspace carve-out (256B aligned)
  size_t off = 0;
  auto alloc = [&](size_t bytes) -> void* {
    off = (off + 255) & ~(size_t)255;
    void* p = (char*)d_ws + off;
    off += bytes;
    return p;
  };
  int4*   meta   = (int4*)  alloc(BB * sizeof(int4));
  bf16_t* nx     = (bf16_t*)alloc((size_t)BB * SS * DD * sizeof(bf16_t));
  bf16_t* wT     = (bf16_t*)alloc((size_t)NE * DD * sizeof(bf16_t));
  bf16_t* qbuf   = (bf16_t*)alloc((size_t)BB * HH * SS * DD * sizeof(bf16_t));
  bf16_t* kbuf   = (bf16_t*)alloc((size_t)BB * HH * SS * DD * sizeof(bf16_t));
  bf16_t* vT     = (bf16_t*)alloc((size_t)BB * HH * DD * SS * sizeof(bf16_t));
  float*  u_last = (float*) alloc((size_t)BB * TH * sizeof(float));
  float*  x_last = (float*) alloc((size_t)BB * DD * sizeof(float));
  float*  o_part = (float*) alloc((size_t)BB * HH * NSPLIT * DD * sizeof(float));
  float*  o_last = (float*) alloc((size_t)BB * TH * sizeof(float));
  float*  hbuf   = (float*) alloc((size_t)BB * DD * sizeof(float));

  prep_kernel<<<1, BB, 0, stream>>>(seq_len, num_tgt, timestamps, meta);
  convert_wT_kernel<<<(NE * DD + 255) / 256, 256, 0, stream>>>(uvqk_w, wT);
  embed_ln_kernel<<<BB * SS, DD, 0, stream>>>(emb_table, pos_emb, time_emb,
                                              ln1_g, ln1_b, item_ids, timestamps,
                                              meta, nx, x_last);
  uvqk_gemm_kernel<<<dim3(NE / 64, (BB * SS / 16) / 4), dim3(32, 4), 0, stream>>>(
      nx, wT, uvqk_b, meta, u_last, qbuf, kbuf, vT);
  attn_kernel<<<dim3(BB * HH, NSPLIT), 32, 0, stream>>>(qbuf, kbuf, vT, meta,
                                                        o_part);
  attn_reduce_kernel<<<(BB * TH + 255) / 256, 256, 0, stream>>>(o_part, o_last);
  tail_kernel<<<BB, DD, 0, stream>>>(o_last, u_last, x_last, ln2_g, ln2_b,
                                     out_w, hbuf);
  head_kernel<<<(VV + 255) / 256, 256, 0, stream>>>(hbuf, head_w, head_b, out);
}
